// SumformerInnerBlock_18674517803376
// MI455X (gfx1250) — matrix operand order repaired and checked by reading, not compiled, not run on hardware
//
#include <hip/hip_runtime.h>
#include <hip/hip_bf16.h>

typedef unsigned short u16;
typedef __attribute__((ext_vector_type(16))) __bf16 bf16x16;
typedef __attribute__((ext_vector_type(8)))  float  floatx8;
typedef __attribute__((ext_vector_type(8)))  unsigned short u16x8;

#define NEG_SLOPE 0.01f
#define ROWS 32           // nodes per block (two 16-row WMMA tiles)
#define LD 528            // LDS activation row stride in bf16 elems (512 + 32B pad, 16B aligned)
#define SIG_LD 272        // sigma LDS stride (256 + pad)

__device__ __forceinline__ u16 f2bf(float f) {
    unsigned u = __builtin_bit_cast(unsigned, f);
    unsigned r = u + 0x7FFFu + ((u >> 16) & 1u);   // round-to-nearest-even
    return (u16)(r >> 16);
}
__device__ __forceinline__ float bf2f(u16 b) {
    return __builtin_bit_cast(float, (unsigned)b << 16);
}
__device__ __forceinline__ float lrelu(float v) {
    return v >= 0.f ? v : NEG_SLOPE * v;
}

union FragU {
    bf16x16 b;
    u16x8   h[2];
};

// ---- single-row-tile GEMM (used by sigma kernel) ----
__device__ __forceinline__ floatx8 wmma_tile(const u16* __restrict__ ldsA, int strideA,
                                             const u16* __restrict__ packedW,
                                             int fan_in, int ct, int lane)
{
    const int r  = lane & 15;
    const int hi = lane >> 4;
    floatx8 acc = {0.f,0.f,0.f,0.f,0.f,0.f,0.f,0.f};
    const int ksteps = fan_in >> 5;
    const u16* wbase = packedW + (size_t)ct * ksteps * 512;
    const u16* arow  = ldsA + r * strideA;
    for (int ks = 0; ks < ksteps; ++ks) {
        FragU a, b;
        a.h[0] = *(const u16x8*)(arow + ks * 32 + 8 * hi);
        a.h[1] = *(const u16x8*)(arow + ks * 32 + 16 + 8 * hi);
        const u16* wp = wbase + (size_t)(ks * 32 + lane) * 16;
        b.h[0] = *(const u16x8*)(wp);
        b.h[1] = *(const u16x8*)(wp + 8);
        acc = __builtin_amdgcn_wmma_f32_16x16x32_bf16(false, a.b, false, b.b,
                                                      (short)0, acc, false, false);
    }
    return acc;
}

// ---- dual-row-tile GEMM: one B-fragment load feeds two independent WMMAs ----
__device__ __forceinline__ void wmma_tile2(const u16* __restrict__ ldsA,
                                           const u16* __restrict__ packedW,
                                           int fan_in, int ct, int lane,
                                           floatx8& acc0, floatx8& acc1)
{
    const int r  = lane & 15;
    const int hi = lane >> 4;
    acc0 = (floatx8){0.f,0.f,0.f,0.f,0.f,0.f,0.f,0.f};
    acc1 = (floatx8){0.f,0.f,0.f,0.f,0.f,0.f,0.f,0.f};
    const int ksteps = fan_in >> 5;
    const u16* wbase = packedW + (size_t)ct * ksteps * 512;
    const u16* arow0 = ldsA + r * LD;          // rows 0..15
    const u16* arow1 = ldsA + (16 + r) * LD;   // rows 16..31
    for (int ks = 0; ks < ksteps; ++ks) {
        FragU a0, a1, b;
        const u16* wp = wbase + (size_t)(ks * 32 + lane) * 16;
        b.h[0]  = *(const u16x8*)(wp);
        b.h[1]  = *(const u16x8*)(wp + 8);
        a0.h[0] = *(const u16x8*)(arow0 + ks * 32 + 8 * hi);
        a0.h[1] = *(const u16x8*)(arow0 + ks * 32 + 16 + 8 * hi);
        a1.h[0] = *(const u16x8*)(arow1 + ks * 32 + 8 * hi);
        a1.h[1] = *(const u16x8*)(arow1 + ks * 32 + 16 + 8 * hi);
        acc0 = __builtin_amdgcn_wmma_f32_16x16x32_bf16(false, a0.b, false, b.b,
                                                       (short)0, acc0, false, false);
        acc1 = __builtin_amdgcn_wmma_f32_16x16x32_bf16(false, a1.b, false, b.b,
                                                       (short)0, acc1, false, false);
    }
}

// Full layer over a 32-node tile: fan_in -> fan_out, bias + optional LeakyReLU, bf16 -> LDS
__device__ __forceinline__ void layer_fwd(const u16* bufIn, u16* bufOut,
                                          const u16* __restrict__ packedW,
                                          const float* __restrict__ bias,
                                          int fan_in, int fan_out, bool act, int tid)
{
    const int lane = tid & 31, wid = tid >> 5;
    const int r = lane & 15, hi = lane >> 4;
    const int tiles_c = fan_out >> 4;
    for (int ct = wid; ct < tiles_c; ct += 8) {
        floatx8 acc0, acc1;
        wmma_tile2(bufIn, packedW, fan_in, ct, lane, acc0, acc1);
        const int col = ct * 16 + r;
        const float bv = bias[col];
        #pragma unroll
        for (int v = 0; v < 8; ++v) {
            float v0 = acc0[v] + bv;
            float v1 = acc1[v] + bv;
            if (act) { v0 = lrelu(v0); v1 = lrelu(v1); }
            bufOut[(v + 8 * hi) * LD + col]      = f2bf(v0);
            bufOut[(16 + v + 8 * hi) * LD + col] = f2bf(v1);
        }
    }
}

// ---------------- prep kernels ----------------
__global__ void zero_f32(float* p, int n) {
    int i = blockIdx.x * blockDim.x + threadIdx.x;
    for (; i < n; i += gridDim.x * blockDim.x) p[i] = 0.f;
}

// Repack fp32 W[fan_in][fan_out] into bf16 WMMA B-fragments:
//   packed[((ct*ksteps + ks)*32 + lane)*16 + j]
__global__ void pack_weights(const float* __restrict__ W, u16* __restrict__ dst,
                             int fan_in, int fan_out)
{
    const int ksteps = fan_in >> 5;
    const int total  = fan_in * fan_out;
    for (int i = blockIdx.x * blockDim.x + threadIdx.x; i < total;
         i += gridDim.x * blockDim.x) {
        const int j    = i & 15;
        const int lane = (i >> 4) & 31;
        const int ks   = (i >> 9) % ksteps;
        const int ct   = i / (ksteps << 9);
        // B layout: lane half h holds K=8h..8h+7 (j<8) and K=16+8h..16+8h+7 (j>=8) of the 32-K step
        const int k = ks * 32 + ((j >> 3) << 4) + ((lane >> 4) << 3) + (j & 7);
        const int c = ct * 16 + (lane & 15);
        dst[i] = f2bf(W[(size_t)k * fan_out + c]);
    }
}

// ---------------- phi + segment sum ----------------
__global__ void __launch_bounds__(256)
phi_kernel(const float* __restrict__ x, const long long* __restrict__ batch, int N,
           const u16* w0, const float* b0, const u16* w1, const float* b1,
           const u16* w2, const float* b2, const u16* w3, const float* b3,
           float* __restrict__ seg_sum, float* __restrict__ counts)
{
    __shared__ __align__(16) u16 bufA[ROWS * LD];
    __shared__ __align__(16) u16 bufB[ROWS * LD];
    __shared__ int gids[ROWS];
    const int tid  = threadIdx.x;
    const int base = blockIdx.x * ROWS;

    for (int i = tid; i < ROWS * 128; i += 256) {
        int m = i >> 7, c = i & 127;
        int node = base + m;
        float v = (node < N) ? x[(size_t)node * 128 + c] : 0.f;
        bufA[m * LD + c] = f2bf(v);
    }
    if (tid < ROWS) {
        int node = base + tid;
        gids[tid] = (node < N) ? (int)batch[node] : -1;
    }
    __syncthreads();

    layer_fwd(bufA, bufB, w0, b0, 128, 512, true, tid); __syncthreads();
    layer_fwd(bufB, bufA, w1, b1, 512, 512, true, tid); __syncthreads();
    layer_fwd(bufA, bufB, w2, b2, 512, 512, true, tid); __syncthreads();
    layer_fwd(bufB, bufA, w3, b3, 512, 256, true, tid); __syncthreads();  // outer LeakyReLU fused

    // segment sum: batch sorted -> run-length flush, one column (of 256) per thread
    {
        float run = 0.f; int gprev = gids[0];
        for (int m = 0; m < ROWS; ++m) {
            int g = gids[m];
            if (g != gprev) {
                if (gprev >= 0) atomicAdd(&seg_sum[gprev * 256 + tid], run);
                run = 0.f; gprev = g;
            }
            run += bf2f(bufA[m * LD + tid]);
        }
        if (gprev >= 0) atomicAdd(&seg_sum[gprev * 256 + tid], run);
    }
    if (tid == 0) {
        float run = 0.f; int gprev = gids[0];
        for (int m = 0; m < ROWS; ++m) {
            int g = gids[m];
            if (g != gprev) {
                if (gprev >= 0) atomicAdd(&counts[gprev], run);
                run = 0.f; gprev = g;
            }
            if (g >= 0) run += 1.f;
        }
        if (gprev >= 0) atomicAdd(&counts[gprev], run);
    }
}

// ---------------- sigma -> sigma_h (64x256 @ 256x512) ----------------
__global__ void __launch_bounds__(256)
sigma_kernel(const float* __restrict__ seg_sum, const float* __restrict__ counts,
             const u16* __restrict__ w_ag, const float* __restrict__ b_ag,
             float* __restrict__ sigma_h)
{
    __shared__ __align__(16) u16 sig[64 * SIG_LD];
    const int tid = threadIdx.x;
    for (int i = tid; i < 64 * 256; i += 256) {
        int g = i >> 8, k = i & 255;
        float c = counts[g]; c = c > 1.f ? c : 1.f;
        sig[g * SIG_LD + k] = f2bf(seg_sum[i] / c);
    }
    __syncthreads();
    const int lane = tid & 31, wid = tid >> 5;
    const int r = lane & 15, hi = lane >> 4;
    for (int t = wid; t < 128; t += 8) {          // 4 row-tiles x 32 col-tiles
        int rt = t >> 5, ct = t & 31;
        floatx8 acc = wmma_tile(sig + rt * 16 * SIG_LD, SIG_LD, w_ag, 256, ct, lane);
        int col = ct * 16 + r;
        float bv = b_ag[col];
        #pragma unroll
        for (int v = 0; v < 8; ++v)
            sigma_h[(size_t)(rt * 16 + v + 8 * hi) * 512 + col] = acc[v] + bv;
    }
}

// ---------------- x@w_in + sigma_h gather + psi MLP ----------------
__global__ void __launch_bounds__(256)
psi_kernel(const float* __restrict__ x, const long long* __restrict__ batch, int N,
           const u16* w_in, const float* b_in, const float* __restrict__ sigma_h,
           const u16* p0, const float* q0, const u16* p1, const float* q1,
           const u16* p2, const float* q2, const u16* p3, const float* q3,
           float* __restrict__ out)
{
    __shared__ __align__(16) u16 bufA[ROWS * LD];
    __shared__ __align__(16) u16 bufB[ROWS * LD];
    __shared__ int gids[ROWS];
    const int tid  = threadIdx.x;
    const int base = blockIdx.x * ROWS;
    const int lane = tid & 31, wid = tid >> 5;
    const int r = lane & 15, hi = lane >> 4;

    for (int i = tid; i < ROWS * 128; i += 256) {
        int m = i >> 7, c = i & 127;
        int node = base + m;
        float v = (node < N) ? x[(size_t)node * 128 + c] : 0.f;
        bufA[m * LD + c] = f2bf(v);
    }
    if (tid < ROWS) {
        int node = base + tid;
        gids[tid] = (node < N) ? (int)batch[node] : -1;
    }
    __syncthreads();

    // input_linear (128->512) fused with sigma_h[batch] add + LeakyReLU
    for (int ct = wid; ct < 32; ct += 8) {
        floatx8 acc0, acc1;
        wmma_tile2(bufA, w_in, 128, ct, lane, acc0, acc1);
        int col = ct * 16 + r;
        float bv = b_in[col];
        #pragma unroll
        for (int v = 0; v < 8; ++v) {
            int m0 = v + 8 * hi;
            int m1 = 16 + v + 8 * hi;
            int g0 = gids[m0];
            int g1 = gids[m1];
            float s0 = (g0 >= 0) ? sigma_h[(size_t)g0 * 512 + col] : 0.f;
            float s1 = (g1 >= 0) ? sigma_h[(size_t)g1 * 512 + col] : 0.f;
            bufB[m0 * LD + col] = f2bf(lrelu(acc0[v] + bv + s0));
            bufB[m1 * LD + col] = f2bf(lrelu(acc1[v] + bv + s1));
        }
    }
    __syncthreads();

    layer_fwd(bufB, bufA, p0, q0, 512, 512, true, tid); __syncthreads();
    layer_fwd(bufA, bufB, p1, q1, 512, 512, true, tid); __syncthreads();
    layer_fwd(bufB, bufA, p2, q2, 512, 512, true, tid); __syncthreads();

    // final 512->128, no activation, straight to global
    if (wid < 8) {
        int ct = wid;
        floatx8 acc0, acc1;
        wmma_tile2(bufA, p3, 512, ct, lane, acc0, acc1);
        int col = ct * 16 + r;
        float bv = q3[col];
        #pragma unroll
        for (int v = 0; v < 8; ++v) {
            int n0 = base + v + 8 * hi;
            int n1 = base + 16 + v + 8 * hi;
            if (n0 < N) out[(size_t)n0 * 128 + col] = acc0[v] + bv;
            if (n1 < N) out[(size_t)n1 * 128 + col] = acc1[v] + bv;
        }
    }
}

extern "C" void kernel_launch(void* const* d_in, const int* in_sizes, int n_in,
                              void* d_out, int out_size, void* d_ws, size_t ws_size,
                              hipStream_t stream)
{
    const float*     x     = (const float*)d_in[0];
    const long long* batch = (const long long*)d_in[1];   // int64 per reference
    const float* phi_w0 = (const float*)d_in[3];  const float* phi_b0 = (const float*)d_in[4];
    const float* phi_w1 = (const float*)d_in[5];  const float* phi_b1 = (const float*)d_in[6];
    const float* phi_w2 = (const float*)d_in[7];  const float* phi_b2 = (const float*)d_in[8];
    const float* phi_w3 = (const float*)d_in[9];  const float* phi_b3 = (const float*)d_in[10];
    const float* psi_w0 = (const float*)d_in[11]; const float* psi_b0 = (const float*)d_in[12];
    const float* psi_w1 = (const float*)d_in[13]; const float* psi_b1 = (const float*)d_in[14];
    const float* psi_w2 = (const float*)d_in[15]; const float* psi_b2 = (const float*)d_in[16];
    const float* psi_w3 = (const float*)d_in[17]; const float* psi_b3 = (const float*)d_in[18];
    const float* w_in   = (const float*)d_in[19]; const float* b_in   = (const float*)d_in[20];
    const float* w_ag   = (const float*)d_in[21]; const float* b_ag   = (const float*)d_in[22];

    const int N = in_sizes[0] / 128;
    float* out = (float*)d_out;

    // workspace layout
    char*  ws   = (char*)d_ws;
    float* seg  = (float*)(ws + 0);          // 64*256 f32 = 65536 B
    float* cnt  = (float*)(ws + 65536);      // 64 f32 (contiguous after seg)
    float* sigh = (float*)(ws + 65792);      // 64*512 f32 = 131072 B
    u16*   pw   = (u16*)(ws + 196864);       // packed bf16 weights
    u16* p_phi0 = pw;                        // 128*512
    u16* p_phi1 = p_phi0 + 65536;            // 512*512
    u16* p_phi2 = p_phi1 + 262144;
    u16* p_phi3 = p_phi2 + 262144;           // 512*256
    u16* p_psi0 = p_phi3 + 131072;
    u16* p_psi1 = p_psi0 + 262144;
    u16* p_psi2 = p_psi1 + 262144;
    u16* p_psi3 = p_psi2 + 262144;           // 512*128
    u16* p_win  = p_psi3 + 65536;            // 128*512
    u16* p_wag  = p_win  + 65536;            // 256*512

    zero_f32<<<65, 256, 0, stream>>>(seg, 64 * 256 + 64);

    pack_weights<<<256, 256, 0, stream>>>(phi_w0, p_phi0, 128, 512);
    pack_weights<<<256, 256, 0, stream>>>(phi_w1, p_phi1, 512, 512);
    pack_weights<<<256, 256, 0, stream>>>(phi_w2, p_phi2, 512, 512);
    pack_weights<<<256, 256, 0, stream>>>(phi_w3, p_phi3, 512, 256);
    pack_weights<<<256, 256, 0, stream>>>(psi_w0, p_psi0, 512, 512);
    pack_weights<<<256, 256, 0, stream>>>(psi_w1, p_psi1, 512, 512);
    pack_weights<<<256, 256, 0, stream>>>(psi_w2, p_psi2, 512, 512);
    pack_weights<<<256, 256, 0, stream>>>(psi_w3, p_psi3, 512, 128);
    pack_weights<<<256, 256, 0, stream>>>(w_in,   p_win,  128, 512);
    pack_weights<<<256, 256, 0, stream>>>(w_ag,   p_wag,  256, 512);

    const int nblk = (N + ROWS - 1) / ROWS;
    phi_kernel<<<nblk, 256, 0, stream>>>(x, batch, N,
        p_phi0, phi_b0, p_phi1, phi_b1, p_phi2, phi_b2, p_phi3, phi_b3, seg, cnt);
    sigma_kernel<<<1, 256, 0, stream>>>(seg, cnt, p_wag, b_ag, sigh);
    psi_kernel<<<nblk, 256, 0, stream>>>(x, batch, N, p_win, b_in, sigh,
        p_psi0, psi_b0, p_psi1, psi_b1, p_psi2, psi_b2, p_psi3, psi_b3, out);
}
